// Local_Module_19138374271325
// MI455X (gfx1250) — matrix operand order repaired and checked by reading, not compiled, not run on hardware
//
#include <hip/hip_runtime.h>

// MPNN layer for MI455X (gfx1250, wave32, WMMA + async-to-LDS staging).
// d = din = 128, C = 256, K = 30 neighbors, D_FF = 512, nodes = B*N = 16384.

#define SLOPE     0.01f
#define LN_EPS    1e-6f
#define MSG_SCALE 30.0f
#define KNB       30

typedef __bf16 bf16_t;
typedef bf16_t v8bf  __attribute__((ext_vector_type(8)));
typedef bf16_t v16bf __attribute__((ext_vector_type(16)));
typedef float  v8f   __attribute__((ext_vector_type(8)));

// ---------------------------------------------------------------------------
// WMMA helpers (CDNA5 V_WMMA_F32_16X16X32_BF16, wave32)
// A 16x32 (M x K): lanes 0-15 row=lane, elems 0..7 = K[8*lh..+8), 8..15 = K[16+8*lh..+8)
// B 32x16 (K x N): lane col = lane&15, elems 0..7 = K[8*lh..+8), 8..15 = K[16+8*lh..+8)
// C/D 16x16 f32:   row = v + 8*(lane>>4), col = lane&15
// ---------------------------------------------------------------------------
__device__ __forceinline__ v16bf load_b_frag(const bf16_t* p) {
  union { v16bf v; v8bf h[2]; } u;
  u.h[0] = *(const v8bf*)(p);
  u.h[1] = *(const v8bf*)(p + 16);
  return u.v;
}

__device__ __forceinline__ v16bf load_a_frag(const bf16_t* p) {
  union { v16bf v; v8bf h[2]; } u;
  u.h[0] = *(const v8bf*)(p);
  u.h[1] = *(const v8bf*)(p + 16);
  return u.v;
}

__device__ __forceinline__ v8f wmma_bf16(v16bf a, v16bf b, v8f c) {
  return __builtin_amdgcn_wmma_f32_16x16x32_bf16(false, a, false, b, (short)0, c,
                                                 false, false);
}

// ---------------------------------------------------------------------------
// Async global->LDS copy, 16 bytes per enabled lane (GLOBAL_LOAD_ASYNC_TO_LDS_B128,
// GV addressing: per-lane 64-bit vaddr).  Tracked with ASYNCcnt.
// Generic pointers to LDS carry the LDS byte offset in their low 32 bits.
// ---------------------------------------------------------------------------
__device__ __forceinline__ void async_copy_b128(const void* gsrc, unsigned lds_off) {
  asm volatile("global_load_async_to_lds_b128 %0, %1, off"
               :: "v"(lds_off), "v"((unsigned long long)gsrc)
               : "memory");
}
__device__ __forceinline__ void wait_async_all() {
  asm volatile("s_wait_asynccnt 0x0" ::: "memory");
}
__device__ __forceinline__ unsigned lds_off32(const void* p) {
  return (unsigned)(unsigned long long)p;
}

// ---------------------------------------------------------------------------
// Workspace layout:
//   bf16 transposed weights at byte 0:
//     W1t  [128][256]  elems      0 .. 32767   (W1t[n][k]  = W1[k*128+n])
//     W2t  [128][128]  elems  32768 .. 49151
//     W3t  [128][128]  elems  49152 .. 65535
//     Wf1t [512][128]  elems  65536 .. 131071  (Wf1t[n][k] = Wf1[k*512+n])
//     Wf2t [128][512]  elems 131072 .. 196607  (Wf2t[n][k] = Wf2[k*128+n])
//   float hVn[16384][128] at byte 393216 (8 MB)
// ---------------------------------------------------------------------------
#define WT_ELEMS    196608
#define HVN_BYTEOFF 393216

__global__ __launch_bounds__(256)
void prep_weights(const float* __restrict__ W1,  const float* __restrict__ W2,
                  const float* __restrict__ W3,  const float* __restrict__ Wf1,
                  const float* __restrict__ Wf2, bf16_t* __restrict__ wt) {
  int t = blockIdx.x * blockDim.x + threadIdx.x;
  if (t >= WT_ELEMS) return;
  float v;
  if (t < 32768) {                      // W1t: n<128, k<256
    int n = t >> 8, k = t & 255;
    v = W1[k * 128 + n];
  } else if (t < 49152) {               // W2t
    int i = t - 32768; int n = i >> 7, k = i & 127;
    v = W2[k * 128 + n];
  } else if (t < 65536) {               // W3t
    int i = t - 49152; int n = i >> 7, k = i & 127;
    v = W3[k * 128 + n];
  } else if (t < 131072) {              // Wf1t: n<512, k<128
    int i = t - 65536; int n = i >> 7, k = i & 127;
    v = Wf1[k * 512 + n];
  } else {                              // Wf2t: n<128, k<512
    int i = t - 131072; int n = i >> 9, k = i & 511;
    v = Wf2[k * 128 + n];
  }
  wt[t] = (bf16_t)v;
}

// ---------------------------------------------------------------------------
// Kernel A: 2 nodes / workgroup (64 edge rows, rows 30..31 of each node = pad)
// message MLP (3 WMMA GEMMs) + attention + masked K-reduction + LN0 -> hVn
// ---------------------------------------------------------------------------
#define EV_STRIDE 264   // 256 + 8 pad (bf16 elems)
#define H_STRIDE  136   // 128 + 8 pad

// dynamic LDS offsets (bytes)
#define A_OFF_EV   0        // bf16 [64][264] = 33792
#define A_OFF_H1   33792    // bf16 [64][136] = 17408
#define A_OFF_H2   51200    // bf16 [64][136] = 17408
#define A_OFF_AV   68608    // f32 [256]      = 1024
#define A_OFF_ATTP 69632    // f32 [64][4]    = 1024
#define A_OFF_E    70656    // f32 [64]       = 256
#define A_OFF_WROW 70912    // f32 [64]       = 256
#define A_OFF_DH   71168    // f32 [2][128]   = 1024
#define A_OFF_RS   72192    // f32 [256]      = 1024
#define A_OFF_RQ   73216    // f32 [256]      = 1024
#define A_OFF_EINV 74240    // f32 [2]
#define A_OFF_WSUM 74248    // f32 [2]
#define A_OFF_RAW  74256    // f32 [64][128]  = 32768 (async h_E staging)
#define A_SMEM     107024

__device__ __forceinline__ void store_act_leaky(bf16_t (*buf)[H_STRIDE], v8f a,
                                                int mbase, int colg, int lh,
                                                float bias) {
#pragma unroll
  for (int v = 0; v < 8; ++v) {
    float x = a[v] + bias;
    x = (x >= 0.f) ? x : SLOPE * x;
    buf[mbase + 8 * lh + v][colg] = (bf16_t)x;
  }
}

__global__ __launch_bounds__(256)
void msg_kernel(const float* __restrict__ h_V, const float* __restrict__ h_E,
                const int* __restrict__ mask_attend,
                const float* __restrict__ b1, const float* __restrict__ b2,
                const float* __restrict__ b3, const float* __restrict__ Avec,
                const float* __restrict__ gain0, const float* __restrict__ beta0,
                const bf16_t* __restrict__ wt, float* __restrict__ hVn) {
  extern __shared__ char smem[];
  bf16_t (*sEV)[EV_STRIDE] = (bf16_t(*)[EV_STRIDE])(smem + A_OFF_EV);
  bf16_t (*sH1)[H_STRIDE]  = (bf16_t(*)[H_STRIDE])(smem + A_OFF_H1);
  bf16_t (*sH2)[H_STRIDE]  = (bf16_t(*)[H_STRIDE])(smem + A_OFF_H2);
  float* sA    = (float*)(smem + A_OFF_AV);
  float (*attp)[4] = (float(*)[4])(smem + A_OFF_ATTP);
  float* ebuf  = (float*)(smem + A_OFF_E);
  float* wrow  = (float*)(smem + A_OFF_WROW);
  float (*dhacc)[128] = (float(*)[128])(smem + A_OFF_DH);
  float* redS  = (float*)(smem + A_OFF_RS);
  float* redQ  = (float*)(smem + A_OFF_RQ);
  float* einv  = (float*)(smem + A_OFF_EINV);
  float* wsum  = (float*)(smem + A_OFF_WSUM);
  float* sRAW  = (float*)(smem + A_OFF_RAW);

  const int t    = threadIdx.x;
  const int wave = t >> 5;
  const int lane = t & 31;
  const int lh   = lane >> 4;     // half-wave select
  const int lc   = lane & 15;     // col / row-in-tile
  const long node0 = (long)blockIdx.x * 2;

  // ---- issue async DMA of h_E rows (raw f32) into sRAW: 2048 x 16B chunks ----
#pragma unroll
  for (int i = 0; i < 8; ++i) {
    int c   = t + 256 * i;          // chunk id
    int row = c >> 5;               // 32 chunks per 512B row
    int k   = row & 31;
    if (k < KNB) {
      int node = row >> 5;
      long gn  = node0 + node;
      const char* g = (const char*)(h_E + ((gn * KNB + k) << 7)) + ((c & 31) << 4);
      unsigned lds  = lds_off32((const char*)sRAW + (row << 9) + ((c & 31) << 4));
      async_copy_b128(g, lds);
    }
  }

  // ---- overlap: small staging while DMA is in flight ----
  sA[t] = Avec[t];
  dhacc[t >> 7][t & 127] = 0.f;
  {
    int row = t >> 2, seg = t & 3;      // 4 threads per row
    int node = row >> 5, k = row & 31;
    long gn  = node0 + node;
    if (k < KNB) {
      if (seg < 2) {
        const float* src = h_V + gn * 128 + seg * 64;
#pragma unroll 8
        for (int i = 0; i < 64; ++i) sEV[row][seg * 64 + i] = (bf16_t)src[i];
      }
    } else {
#pragma unroll 8
      for (int i = 0; i < 64; ++i) sEV[row][seg * 64 + i] = (bf16_t)0.f;
    }
  }

  wait_async_all();
  __syncthreads();

  // ---- convert staged h_E f32 -> bf16 into sEV[:,128:256] ----
  {
    int row = t >> 2, seg = t & 3;
    if ((row & 31) < KNB) {
      const float4* src = (const float4*)(sRAW + (row << 7) + seg * 32);
#pragma unroll
      for (int j = 0; j < 8; ++j) {
        float4 x = src[j];
        int c = 128 + seg * 32 + j * 4;
        sEV[row][c + 0] = (bf16_t)x.x;
        sEV[row][c + 1] = (bf16_t)x.y;
        sEV[row][c + 2] = (bf16_t)x.z;
        sEV[row][c + 3] = (bf16_t)x.w;
      }
    }
  }
  __syncthreads();

  // ---- attention: w[row] = mask * exp(sigmoid(leaky(hEV . A))) / (sum * SCALE)
  {
    int row = t >> 2, seg = t & 3;
    float p = 0.f;
#pragma unroll 8
    for (int i = 0; i < 64; ++i) {
      int c = seg * 64 + i;
      p += (float)sEV[row][c] * sA[c];
    }
    attp[row][seg] = p;
  }
  __syncthreads();
  if (t < 64) {
    float lg = attp[t][0] + attp[t][1] + attp[t][2] + attp[t][3];
    lg = (lg >= 0.f) ? lg : SLOPE * lg;
    float s = 1.f / (1.f + expf(-lg));
    ebuf[t] = expf(s);
  }
  __syncthreads();
  if (t < 2) {
    float se = 0.f;
    for (int k = 0; k < KNB; ++k) se += ebuf[t * 32 + k];
    einv[t] = 1.f / (se * MSG_SCALE);
  }
  __syncthreads();
  if (t < 64) {
    int node = t >> 5, k = t & 31;
    float wv = 0.f;
    if (k < KNB) {
      long gn = node0 + node;
      wv = (float)mask_attend[gn * KNB + k] * ebuf[t] * einv[node];
    }
    wrow[t] = wv;
  }
  __syncthreads();
  if (t < 2) {
    float s = 0.f;
    for (int i = 0; i < 32; ++i) s += wrow[t * 32 + i];
    wsum[t] = s;   // read only after later barriers
  }

  const bf16_t* W1t = wt;
  const bf16_t* W2t = wt + 32768;
  const bf16_t* W3t = wt + 49152;
  const int colg = wave * 16 + lc;   // each wave owns one 16-col tile (N=128)

  // ---- GEMM1: H1 = leaky(hEV[64x256] @ W1 + b1) ----
  {
    const bf16_t* Bp = W1t + (long)colg * 256 + 8 * lh;
    v8f acc0 = {}, acc1 = {}, acc2 = {}, acc3 = {};
#pragma unroll
    for (int ks = 0; ks < 8; ++ks) {
      const int k0 = ks * 32;
      v16bf bf = load_b_frag(Bp + k0);
      v16bf a0 = load_a_frag(&sEV[ 0 + lc][k0 + 8 * lh]);
      v16bf a1 = load_a_frag(&sEV[16 + lc][k0 + 8 * lh]);
      v16bf a2 = load_a_frag(&sEV[32 + lc][k0 + 8 * lh]);
      v16bf a3 = load_a_frag(&sEV[48 + lc][k0 + 8 * lh]);
      acc0 = wmma_bf16(a0, bf, acc0);
      acc1 = wmma_bf16(a1, bf, acc1);
      acc2 = wmma_bf16(a2, bf, acc2);
      acc3 = wmma_bf16(a3, bf, acc3);
    }
    const float bias = b1[colg];
    store_act_leaky(sH1, acc0,  0, colg, lh, bias);
    store_act_leaky(sH1, acc1, 16, colg, lh, bias);
    store_act_leaky(sH1, acc2, 32, colg, lh, bias);
    store_act_leaky(sH1, acc3, 48, colg, lh, bias);
  }
  __syncthreads();

  // ---- GEMM2: H2 = leaky(H1[64x128] @ W2 + b2) ----
  {
    const bf16_t* Bp = W2t + (long)colg * 128 + 8 * lh;
    v8f acc0 = {}, acc1 = {}, acc2 = {}, acc3 = {};
#pragma unroll
    for (int ks = 0; ks < 4; ++ks) {
      const int k0 = ks * 32;
      v16bf bf = load_b_frag(Bp + k0);
      v16bf a0 = load_a_frag(&sH1[ 0 + lc][k0 + 8 * lh]);
      v16bf a1 = load_a_frag(&sH1[16 + lc][k0 + 8 * lh]);
      v16bf a2 = load_a_frag(&sH1[32 + lc][k0 + 8 * lh]);
      v16bf a3 = load_a_frag(&sH1[48 + lc][k0 + 8 * lh]);
      acc0 = wmma_bf16(a0, bf, acc0);
      acc1 = wmma_bf16(a1, bf, acc1);
      acc2 = wmma_bf16(a2, bf, acc2);
      acc3 = wmma_bf16(a3, bf, acc3);
    }
    const float bias = b2[colg];
    store_act_leaky(sH2, acc0,  0, colg, lh, bias);
    store_act_leaky(sH2, acc1, 16, colg, lh, bias);
    store_act_leaky(sH2, acc2, 32, colg, lh, bias);
    store_act_leaky(sH2, acc3, 48, colg, lh, bias);
  }
  __syncthreads();

  // ---- GEMM3: M = H2 @ W3 ; dh[node] += sum_rows w[row]*M[row]  (b3 via wsum) ----
  {
    const bf16_t* Bp = W3t + (long)colg * 128 + 8 * lh;
    v8f acc0 = {}, acc1 = {}, acc2 = {}, acc3 = {};
#pragma unroll
    for (int ks = 0; ks < 4; ++ks) {
      const int k0 = ks * 32;
      v16bf bf = load_b_frag(Bp + k0);
      v16bf a0 = load_a_frag(&sH2[ 0 + lc][k0 + 8 * lh]);
      v16bf a1 = load_a_frag(&sH2[16 + lc][k0 + 8 * lh]);
      v16bf a2 = load_a_frag(&sH2[32 + lc][k0 + 8 * lh]);
      v16bf a3 = load_a_frag(&sH2[48 + lc][k0 + 8 * lh]);
      acc0 = wmma_bf16(a0, bf, acc0);
      acc1 = wmma_bf16(a1, bf, acc1);
      acc2 = wmma_bf16(a2, bf, acc2);
      acc3 = wmma_bf16(a3, bf, acc3);
    }
    float p0 = 0.f, p1 = 0.f, p2 = 0.f, p3 = 0.f;
#pragma unroll
    for (int v = 0; v < 8; ++v) {
      p0 += acc0[v] * wrow[ 0 + 8 * lh + v];
      p1 += acc1[v] * wrow[16 + 8 * lh + v];
      p2 += acc2[v] * wrow[32 + 8 * lh + v];
      p3 += acc3[v] * wrow[48 + 8 * lh + v];
    }
    atomicAdd(&dhacc[0][colg], p0 + p1);
    atomicAdd(&dhacc[1][colg], p2 + p3);
  }
  __syncthreads();

  // ---- residual + LayerNorm0 -> hVn ----
  {
    int node = t >> 7, c = t & 127;
    long gn = node0 + node;
    float x = h_V[gn * 128 + c] + dhacc[node][c] + b3[c] * wsum[node];
    redS[t] = x;
    redQ[t] = x * x;
    __syncthreads();
#pragma unroll
    for (int s = 64; s > 0; s >>= 1) {
      if ((t & 127) < s) { redS[t] += redS[t + s]; redQ[t] += redQ[t + s]; }
      __syncthreads();
    }
    float mu  = redS[node * 128] * (1.f / 128.f);
    float var = redQ[node * 128] * (1.f / 128.f) - mu * mu;
    hVn[gn * 128 + c] = gain0[c] * (x - mu) * rsqrtf(var + LN_EPS) + beta0[c];
  }
}

// ---------------------------------------------------------------------------
// Kernel B: FFN (128 -> 512 relu -> 128) + residual + LN1 + node mask.
// 64 nodes / workgroup.  hVn tile staged via async DMA into sY (raw f32),
// which doubles as the residual source and later holds the pre-LN output.
// ---------------------------------------------------------------------------
#define F_STRIDE 520   // 512 + 8 pad
#define B_OFF_F   0        // bf16 [64][520] = 66560
#define B_OFF_X   66560    // bf16 [64][136] = 17408
#define B_OFF_Y   83968    // f32  [64][128] = 32768
#define B_OFF_RS  116736   // f32  [256]
#define B_OFF_RQ  117760   // f32  [256]
#define B_SMEM    118784

__global__ __launch_bounds__(256)
void ffn_kernel(const float* __restrict__ hVn, const int* __restrict__ mask_V,
                const float* __restrict__ bf1v, const float* __restrict__ bf2v,
                const float* __restrict__ gain1, const float* __restrict__ beta1,
                const bf16_t* __restrict__ wt, float* __restrict__ out) {
  extern __shared__ char smem[];
  bf16_t (*sF)[F_STRIDE] = (bf16_t(*)[F_STRIDE])(smem + B_OFF_F);
  bf16_t (*sX)[H_STRIDE] = (bf16_t(*)[H_STRIDE])(smem + B_OFF_X);
  float  (*sY)[128]      = (float(*)[128])(smem + B_OFF_Y);
  float* redS = (float*)(smem + B_OFF_RS);
  float* redQ = (float*)(smem + B_OFF_RQ);

  const int t    = threadIdx.x;
  const int wave = t >> 5;
  const int lane = t & 31;
  const int lh   = lane >> 4;
  const int lc   = lane & 15;
  const long gbase = (long)blockIdx.x * 64;

  // ---- async DMA hVn tile (64x128 f32 = 32 KB) into sY ----
#pragma unroll
  for (int i = 0; i < 8; ++i) {
    int c = t + 256 * i;                // 2048 x 16B chunks
    const char* g = (const char*)(hVn + gbase * 128) + ((long)c << 4);
    unsigned lds  = lds_off32((const char*)sY + (c << 4));
    async_copy_b128(g, lds);
  }
  wait_async_all();
  __syncthreads();

  // convert to bf16 for WMMA A-matrix
  {
    int row = t >> 2, seg = t & 3;
    const float4* src = (const float4*)((const float*)sY + (row << 7) + seg * 32);
#pragma unroll
    for (int j = 0; j < 8; ++j) {
      float4 x = src[j];
      int c = seg * 32 + j * 4;
      sX[row][c + 0] = (bf16_t)x.x;
      sX[row][c + 1] = (bf16_t)x.y;
      sX[row][c + 2] = (bf16_t)x.z;
      sX[row][c + 3] = (bf16_t)x.w;
    }
  }
  __syncthreads();

  const bf16_t* Wf1t = wt + 65536;    // [512][128]
  const bf16_t* Wf2t = wt + 131072;   // [128][512]

  // ---- F1: H = relu(hVn[64x128] @ Wf1 + bf1), N=512 (4 col-tiles per wave) ----
  for (int q = 0; q < 4; ++q) {
    const int nt   = wave * 4 + q;
    const int colg = nt * 16 + lc;
    const bf16_t* Bp = Wf1t + (long)colg * 128 + 8 * lh;
    v8f acc0 = {}, acc1 = {}, acc2 = {}, acc3 = {};
#pragma unroll
    for (int ks = 0; ks < 4; ++ks) {
      const int k0 = ks * 32;
      v16bf bf = load_b_frag(Bp + k0);
      v16bf a0 = load_a_frag(&sX[ 0 + lc][k0 + 8 * lh]);
      v16bf a1 = load_a_frag(&sX[16 + lc][k0 + 8 * lh]);
      v16bf a2 = load_a_frag(&sX[32 + lc][k0 + 8 * lh]);
      v16bf a3 = load_a_frag(&sX[48 + lc][k0 + 8 * lh]);
      acc0 = wmma_bf16(a0, bf, acc0);
      acc1 = wmma_bf16(a1, bf, acc1);
      acc2 = wmma_bf16(a2, bf, acc2);
      acc3 = wmma_bf16(a3, bf, acc3);
    }
    const float bias = bf1v[colg];
#pragma unroll
    for (int v = 0; v < 8; ++v) {
      float x0 = acc0[v] + bias; sF[ 0 + 8 * lh + v][colg] = (bf16_t)fmaxf(x0, 0.f);
      float x1 = acc1[v] + bias; sF[16 + 8 * lh + v][colg] = (bf16_t)fmaxf(x1, 0.f);
      float x2 = acc2[v] + bias; sF[32 + 8 * lh + v][colg] = (bf16_t)fmaxf(x2, 0.f);
      float x3 = acc3[v] + bias; sF[48 + 8 * lh + v][colg] = (bf16_t)fmaxf(x3, 0.f);
    }
  }
  __syncthreads();

  // ---- F2: Y = hVn + H[64x512] @ Wf2 + bf2, N=128 (1 col-tile per wave) ----
  {
    const int colg = wave * 16 + lc;
    const bf16_t* Bp = Wf2t + (long)colg * 512 + 8 * lh;
    v8f acc0 = {}, acc1 = {}, acc2 = {}, acc3 = {};
#pragma unroll
    for (int ks = 0; ks < 16; ++ks) {
      const int k0 = ks * 32;
      v16bf bf = load_b_frag(Bp + k0);
      v16bf a0 = load_a_frag(&sF[ 0 + lc][k0 + 8 * lh]);
      v16bf a1 = load_a_frag(&sF[16 + lc][k0 + 8 * lh]);
      v16bf a2 = load_a_frag(&sF[32 + lc][k0 + 8 * lh]);
      v16bf a3 = load_a_frag(&sF[48 + lc][k0 + 8 * lh]);
      acc0 = wmma_bf16(a0, bf, acc0);
      acc1 = wmma_bf16(a1, bf, acc1);
      acc2 = wmma_bf16(a2, bf, acc2);
      acc3 = wmma_bf16(a3, bf, acc3);
    }
    const float bias = bf2v[colg];
#pragma unroll
    for (int v = 0; v < 8; ++v) {
      int r0 =  0 + 8 * lh + v;
      int r1 = 16 + 8 * lh + v;
      int r2 = 32 + 8 * lh + v;
      int r3 = 48 + 8 * lh + v;
      sY[r0][colg] = acc0[v] + bias + sY[r0][colg];   // residual from staged tile
      sY[r1][colg] = acc1[v] + bias + sY[r1][colg];
      sY[r2][colg] = acc2[v] + bias + sY[r2][colg];
      sY[r3][colg] = acc3[v] + bias + sY[r3][colg];
    }
  }
  __syncthreads();

  // ---- LayerNorm1 + node mask ----
  {
    int node = t >> 2, seg = t & 3;   // 4 threads per node
    float s = 0.f, q2 = 0.f;
#pragma unroll 8
    for (int i = 0; i < 32; ++i) {
      float x = sY[node][seg * 32 + i];
      s += x; q2 += x * x;
    }
    redS[t] = s; redQ[t] = q2;
    __syncthreads();
    if (seg == 0) {
      float su = redS[t] + redS[t + 1] + redS[t + 2] + redS[t + 3];
      float sq = redQ[t] + redQ[t + 1] + redQ[t + 2] + redQ[t + 3];
      float mu = su * (1.f / 128.f);
      redS[t] = mu;
      redQ[t] = rsqrtf(sq * (1.f / 128.f) - mu * mu + LN_EPS);
    }
    __syncthreads();
    float mu = redS[node * 4], rs = redQ[node * 4];
    float mv = (float)mask_V[gbase + node];
#pragma unroll 8
    for (int i = 0; i < 32; ++i) {
      int c = seg * 32 + i;
      float x = sY[node][c];
      out[(gbase + node) * 128 + c] = mv * (gain1[c] * (x - mu) * rs + beta1[c]);
    }
  }
}

// ---------------------------------------------------------------------------
extern "C" void kernel_launch(void* const* d_in, const int* in_sizes, int n_in,
                              void* d_out, int out_size, void* d_ws, size_t ws_size,
                              hipStream_t stream) {
  const float* h_V      = (const float*)d_in[0];
  const float* h_E      = (const float*)d_in[1];
  const int*   mask_V   = (const int*)d_in[2];
  const int*   mask_att = (const int*)d_in[3];
  const float* W1  = (const float*)d_in[4];
  const float* b1  = (const float*)d_in[5];
  const float* W2  = (const float*)d_in[6];
  const float* b2  = (const float*)d_in[7];
  const float* W3  = (const float*)d_in[8];
  const float* b3  = (const float*)d_in[9];
  const float* Av  = (const float*)d_in[10];
  const float* g0  = (const float*)d_in[11];
  const float* be0 = (const float*)d_in[12];
  const float* g1  = (const float*)d_in[13];
  const float* be1 = (const float*)d_in[14];
  const float* Wf1 = (const float*)d_in[15];
  const float* bf1 = (const float*)d_in[16];
  const float* Wf2 = (const float*)d_in[17];
  const float* bf2 = (const float*)d_in[18];
  float* out = (float*)d_out;

  bf16_t* wt = (bf16_t*)d_ws;
  float* hVn = (float*)((char*)d_ws + HVN_BYTEOFF);

  const int BN = in_sizes[0] / 128;   // 16384 nodes

  prep_weights<<<(WT_ELEMS + 255) / 256, 256, 0, stream>>>(W1, W2, W3, Wf1, Wf2, wt);
  msg_kernel<<<BN / 2, 256, A_SMEM, stream>>>(h_V, h_E, mask_att, b1, b2, b3, Av,
                                              g0, be0, wt, hVn);
  ffn_kernel<<<BN / 64, 256, B_SMEM, stream>>>(hVn, mask_V, bf1, bf2, g1, be1,
                                               wt, out);
}
